// Block_56538949484919
// MI455X (gfx1250) — matrix operand order repaired and checked by reading, not compiled, not run on hardware
//
#include <hip/hip_runtime.h>
#include <hip/hip_bf16.h>
#include <cmath>

// Problem constants (reference: B=4, L=2048, C=1024, H=16)
#define BB 4
#define LL 2048
#define CC 1024
#define HH 16
#define HDIM 64
#define MM (BB * LL)   // 8192 rows
#define FCDIM (4 * CC) // 4096

typedef __attribute__((ext_vector_type(16))) __bf16 v16bf;
typedef __attribute__((ext_vector_type(8)))  float  v8f;

union BFrag { v16bf v; unsigned int d[8]; };

__device__ __forceinline__ unsigned short f2bf(float f) {
  unsigned int u = __float_as_uint(f);
  u += 0x7FFFu + ((u >> 16) & 1u);   // round-to-nearest-even
  return (unsigned short)(u >> 16);
}

// Generic pointers to LDS carry the LDS byte address in the low 32 bits
// (ISA 10.2 aperture mapping).
__device__ __forceinline__ unsigned int lds_addr(const void* p) {
  return (unsigned int)(unsigned long long)p;
}

// CDNA5 async memory->LDS copy, 16B per lane, tracked by ASYNCcnt.
__device__ __forceinline__ void async_copy_b128(unsigned int lds, const void* gaddr) {
  asm volatile("global_load_async_to_lds_b128 %0, %1, off"
               :: "v"(lds), "v"((unsigned long long)gaddr)
               : "memory");
}
__device__ __forceinline__ void async_wait0() {
  asm volatile("s_wait_asynccnt 0x0" ::: "memory");
}

// A-matrix 16x32 bf16 fragment (ISA 7.12.2): lane m = L%16, half = L/16.
// VGPR v<4 -> K = 2v + 8*half ; v>=4 -> K = 16 + 2(v-4) + 8*half.
__device__ __forceinline__ v16bf load_a_frag(const unsigned short* base, int ld, int lane) {
  const int lhalf = lane >> 4, m = lane & 15;
  const unsigned short* row = base + m * ld;
  BFrag f;
#pragma unroll
  for (int v = 0; v < 8; ++v) {
    const int k = (v < 4) ? (2 * v + 8 * lhalf) : (16 + 2 * (v - 4) + 8 * lhalf);
    f.d[v] = *(const unsigned int*)(row + k);
  }
  return f.v;
}

// B-matrix 32x16 bf16 fragment: lane n = L%16, half = L/16; VGPR v<4 -> K = 2v + 8*half ; v>=4 -> K = 16 + 2(v-4) + 8*half.
// Tiles are staged N-major in LDS ([n][k]) so each VGPR is one aligned dword load.
__device__ __forceinline__ v16bf load_b_frag(const unsigned short* base, int ld, int lane) {
  const int lhalf = lane >> 4, n = lane & 15;
  const unsigned short* row = base + n * ld;
  BFrag f;
#pragma unroll
  for (int v = 0; v < 8; ++v) {
    const int k = (v < 4) ? (2 * v + 8 * lhalf) : (16 + 2 * (v - 4) + 8 * lhalf);
    f.d[v] = *(const unsigned int*)(row + k);
  }
  return f.v;
}

__device__ __forceinline__ v8f wmma_bf16(v16bf a, v16bf b, v8f c) {
  return __builtin_amdgcn_wmma_f32_16x16x32_bf16(false, a, false, b, (short)0, c, false, false);
}

__device__ __forceinline__ float gelu_tanh(float x) {
  const float k0 = 0.7978845608028654f, k1 = 0.044715f;
  return 0.5f * x * (1.0f + tanhf(k0 * (x + k1 * x * x * x)));
}

// ---------------------------------------------------------------------------
// f32 [K,N] -> bf16 transposed [N,K] (weights become N-major for the GEMM)
__global__ __launch_bounds__(256) void cvt_f32_bf16_t(const float* __restrict__ s,
                                                      unsigned short* __restrict__ d,
                                                      int K, int N) {
  int i = blockIdx.x * 256 + threadIdx.x;
  if (i < K * N) {
    int k = i / N, n = i % N;
    d[(size_t)n * K + k] = f2bf(s[i]);
  }
}

// ---------------------------------------------------------------------------
// LayerNorm over C=1024, one row per 256-thread block, bf16 output
__global__ __launch_bounds__(256) void layernorm_to_bf16(
    const float* __restrict__ x, const float* __restrict__ w,
    const float* __restrict__ b, unsigned short* __restrict__ out) {
  __shared__ float s1[256], s2[256];
  const int row = blockIdx.x, t = threadIdx.x;
  const float* xr = x + (size_t)row * CC;
  float v0 = xr[t], v1 = xr[t + 256], v2 = xr[t + 512], v3 = xr[t + 768];
  s1[t] = v0 + v1 + v2 + v3;
  s2[t] = v0 * v0 + v1 * v1 + v2 * v2 + v3 * v3;
  __syncthreads();
  for (int off = 128; off > 0; off >>= 1) {
    if (t < off) { s1[t] += s1[t + off]; s2[t] += s2[t + off]; }
    __syncthreads();
  }
  const float mean = s1[0] * (1.0f / CC);
  const float var  = s2[0] * (1.0f / CC) - mean * mean;
  const float rstd = rsqrtf(var + 1e-5f);
  unsigned short* orow = out + (size_t)row * CC;
  orow[t]       = f2bf((v0 - mean) * rstd * w[t]       + b[t]);
  orow[t + 256] = f2bf((v1 - mean) * rstd * w[t + 256] + b[t + 256]);
  orow[t + 512] = f2bf((v2 - mean) * rstd * w[t + 512] + b[t + 512]);
  orow[t + 768] = f2bf((v3 - mean) * rstd * w[t + 768] + b[t + 768]);
}

// ---------------------------------------------------------------------------
// WMMA GEMM: C[M,N] = A[M,K](bf16 row-major) x Bt[N,K](bf16 N-major) + bias.
// Block tile 64x256, 8 waves (2x4), each wave 32x64 = 8 WMMA per 32-deep K step.
// Double-buffered LDS tiles filled by GLOBAL_LOAD_ASYNC_TO_LDS_B128 (ASYNCcnt).
// EPI: 0 = bf16 ; 1 = gelu->bf16 ; 2 = f32 + f32 residual ;
//      3 = bf16 scattered per-head transpose [B,H,HD,L] (for V).
template <int EPI>
__global__ __launch_bounds__(256) void gemm_bf16_wmma(
    const unsigned short* __restrict__ A, const unsigned short* __restrict__ Bt,
    const float* __restrict__ bias, const float* __restrict__ res,
    unsigned short* __restrict__ outb, float* __restrict__ outf,
    int M, int N, int K) {
  __shared__ __align__(16) unsigned short Asub[2][64 * 32];    //  2 x 4 KB
  __shared__ __align__(16) unsigned short Bsub[2][256 * 32];   //  2 x 16 KB
  const int t = threadIdx.x;
  const int lane = t & 31, wave = t >> 5;
  const int wm = wave >> 2, wn = wave & 3;
  const int m0 = blockIdx.y * 64, n0 = blockIdx.x * 256;
  const v8f vzero = {0, 0, 0, 0, 0, 0, 0, 0};

  v8f acc[2][4];
#pragma unroll
  for (int i = 0; i < 2; ++i)
#pragma unroll
    for (int j = 0; j < 4; ++j) acc[i][j] = vzero;

  const int T = K >> 5;

  // async-stage one 64x32 A tile + 256x32 B tile (all contiguous b128 rows)
  auto stage = [&](int kt, int buf) {
    {
      int r = t >> 2, seg = (t & 3) * 8;
      async_copy_b128(lds_addr(&Asub[buf][r * 32 + seg]),
                      A + (size_t)(m0 + r) * K + kt * 32 + seg);
    }
#pragma unroll
    for (int i = 0; i < 4; ++i) {
      int lin = t + i * 256;
      int n = lin >> 2, seg = (lin & 3) * 8;
      async_copy_b128(lds_addr(&Bsub[buf][n * 32 + seg]),
                      Bt + (size_t)(n0 + n) * K + kt * 32 + seg);
    }
  };

  stage(0, 0);
  async_wait0();
  __syncthreads();

  for (int kt = 0; kt < T; ++kt) {
    if (kt + 1 < T) stage(kt + 1, (kt + 1) & 1);   // fill other buffer async

    const unsigned short* As = Asub[kt & 1];
    const unsigned short* Bs = Bsub[kt & 1];
    v16bf af[2], bfr[4];
#pragma unroll
    for (int mi = 0; mi < 2; ++mi)
      af[mi] = load_a_frag(As + (wm * 32 + mi * 16) * 32, 32, lane);
#pragma unroll
    for (int ni = 0; ni < 4; ++ni)
      bfr[ni] = load_b_frag(Bs + (wn * 64 + ni * 16) * 32, 32, lane);
#pragma unroll
    for (int mi = 0; mi < 2; ++mi)
#pragma unroll
      for (int ni = 0; ni < 4; ++ni)
        acc[mi][ni] = wmma_bf16(af[mi], bfr[ni], acc[mi][ni]);

    async_wait0();
    __syncthreads();
  }

  // Epilogue: C/D frag layout -> row = r + 8*(lane/16), col = lane%16
  const int lhalf = lane >> 4, nl = lane & 15;
#pragma unroll
  for (int mi = 0; mi < 2; ++mi)
#pragma unroll
    for (int ni = 0; ni < 4; ++ni) {
      const int col = n0 + wn * 64 + ni * 16 + nl;
      const float bv = bias[col];
#pragma unroll
      for (int r = 0; r < 8; ++r) {
        const int row = m0 + wm * 32 + mi * 16 + r + 8 * lhalf;
        float x = acc[mi][ni][r] + bv;
        if constexpr (EPI == 0) {
          outb[(size_t)row * N + col] = f2bf(x);
        } else if constexpr (EPI == 1) {
          outb[(size_t)row * N + col] = f2bf(gelu_tanh(x));
        } else if constexpr (EPI == 2) {
          outf[(size_t)row * N + col] = x + res[(size_t)row * N + col];
        } else {
          // V^T layout [B,H,HD,L]: b=row/L, l=row%L, h=col/HD, d=col%HD
          const size_t idx =
              (((size_t)((row >> 11) * HH + (col >> 6)) * HDIM) + (col & 63)) * LL +
              (row & (LL - 1));
          outb[idx] = f2bf(x);
        }
      }
    }
}

// ---------------------------------------------------------------------------
// Attention: per (b,h) and 64-query-row block, stream 64-key tiles.
// S = Q K^T * 1/8 ; P = exp(S) (unnormalized, f32 row sums) ; Y = (P V) / l.
// Q/K are [B,L,C] bf16 (head h at cols h*64..); Vt is [B,H,HD,L] bf16.
// K / Vt tiles double-buffered via async LDS loads.
__global__ __launch_bounds__(256) void attn_wmma(
    const unsigned short* __restrict__ Q, const unsigned short* __restrict__ Kd,
    const unsigned short* __restrict__ Vt, unsigned short* __restrict__ Y) {
  __shared__ __align__(16) unsigned short Qs[64 * 64];        // 8 KB
  __shared__ __align__(16) unsigned short Ks[2][64 * 64];     // 16 KB, [j][d]
  __shared__ __align__(16) unsigned short Vts[2][64 * 64];    // 16 KB, [d][j]
  __shared__ __align__(16) unsigned short Ps[64 * 64];        // 8 KB
  __shared__ float lsum[64];

  const int t = threadIdx.x, lane = t & 31, wave = t >> 5;
  const int wm = wave >> 2, wn = wave & 3;
  const int rb = blockIdx.x;                 // query row block 0..31
  const int bh = blockIdx.y;                 // 0..63
  const int bidx = bh >> 4;
  const int h = bh & 15;
  const size_t rowBase = (size_t)bidx * LL;
  const int colBase = h * HDIM;
  const int lhalf = lane >> 4, nl = lane & 15;
  const v8f vzero = {0, 0, 0, 0, 0, 0, 0, 0};

  // async copy of one 64x64 bf16 tile (row stride in elements)
  auto stage64 = [&](const unsigned short* g, size_t stride, unsigned short* s) {
#pragma unroll
    for (int i = 0; i < 2; ++i) {
      int lin = t + i * 256;
      int r = lin >> 3, seg = (lin & 7) * 8;
      async_copy_b128(lds_addr(s + r * 64 + seg), g + (size_t)r * stride + seg);
    }
  };

  const unsigned short* qg  = Q  + (rowBase + rb * 64) * CC + colBase;
  const unsigned short* vtg = Vt + (size_t)bh * HDIM * LL;

  stage64(qg, CC, Qs);
  stage64(Kd + (rowBase + 0) * CC + colBase, CC, Ks[0]);
  stage64(vtg + 0, LL, Vts[0]);
  if (t < 64) lsum[t] = 0.0f;

  v8f yacc[2];
  yacc[0] = vzero;
  yacc[1] = vzero;

  async_wait0();
  __syncthreads();

  for (int kt = 0; kt < LL / 64; ++kt) {
    if (kt + 1 < LL / 64) {
      const int nb = (kt + 1) & 1;
      stage64(Kd + (rowBase + (kt + 1) * 64) * CC + colBase, CC, Ks[nb]);
      stage64(vtg + (kt + 1) * 64, LL, Vts[nb]);
    }
    const unsigned short* Kb = Ks[kt & 1];
    const unsigned short* Vb = Vts[kt & 1];

    // S subtiles: wave covers rows wm*32+mi*16, cols wn*16; K-depth 64 = 2 WMMA
#pragma unroll
    for (int mi = 0; mi < 2; ++mi) {
      v8f s = vzero;
      const unsigned short* arow = Qs + (wm * 32 + mi * 16) * 64;
      const unsigned short* brow = Kb + (wn * 16) * 64;
      s = wmma_bf16(load_a_frag(arow, 64, lane),      load_b_frag(brow, 64, lane),      s);
      s = wmma_bf16(load_a_frag(arow + 32, 64, lane), load_b_frag(brow + 32, 64, lane), s);
#pragma unroll
      for (int r = 0; r < 8; ++r) {
        const int rl = wm * 32 + mi * 16 + r + 8 * lhalf;
        const int cl = wn * 16 + nl;
        float e = __expf(s[r] * 0.125f);     // 1/sqrt(64)
        Ps[rl * 64 + cl] = f2bf(e);
        float ss = e;                         // row-sum across 16 lanes of a half
        ss += __shfl_xor(ss, 1, 16);
        ss += __shfl_xor(ss, 2, 16);
        ss += __shfl_xor(ss, 4, 16);
        ss += __shfl_xor(ss, 8, 16);
        if (nl == 0) atomicAdd(&lsum[rl], ss);
      }
    }
    __syncthreads();

    // Y += P V
#pragma unroll
    for (int mi = 0; mi < 2; ++mi) {
      const unsigned short* arow = Ps + (wm * 32 + mi * 16) * 64;
      const unsigned short* brow = Vb + (wn * 16) * 64;
      yacc[mi] = wmma_bf16(load_a_frag(arow, 64, lane),      load_b_frag(brow, 64, lane),      yacc[mi]);
      yacc[mi] = wmma_bf16(load_a_frag(arow + 32, 64, lane), load_b_frag(brow + 32, 64, lane), yacc[mi]);
    }
    async_wait0();
    __syncthreads();
  }

  // Normalize and store (Y is [B,L,C] bf16)
#pragma unroll
  for (int mi = 0; mi < 2; ++mi)
#pragma unroll
    for (int r = 0; r < 8; ++r) {
      const int rl = wm * 32 + mi * 16 + r + 8 * lhalf;
      const int cl = wn * 16 + nl;
      float y = yacc[mi][r] / lsum[rl];
      Y[(rowBase + rb * 64 + rl) * CC + colBase + cl] = f2bf(y);
    }
}

// ---------------------------------------------------------------------------
extern "C" void kernel_launch(void* const* d_in, const int* in_sizes, int n_in,
                              void* d_out, int out_size, void* d_ws, size_t ws_size,
                              hipStream_t stream) {
  (void)in_sizes; (void)n_in; (void)out_size; (void)ws_size;
  const float* h_private = (const float*)d_in[0];
  const float* h_shared  = (const float*)d_in[1];
  const float* ln1p_w = (const float*)d_in[2];
  const float* ln1p_b = (const float*)d_in[3];
  const float* ln1s_w = (const float*)d_in[4];
  const float* ln1s_b = (const float*)d_in[5];
  const float* Wq  = (const float*)d_in[6];  const float* bq   = (const float*)d_in[7];
  const float* Wk  = (const float*)d_in[8];  const float* bk   = (const float*)d_in[9];
  const float* Wv  = (const float*)d_in[10]; const float* bv   = (const float*)d_in[11];
  const float* Wvt = (const float*)d_in[12]; const float* bvt  = (const float*)d_in[13];
  const float* Wcp = (const float*)d_in[14]; const float* bcp  = (const float*)d_in[15];
  const float* ln2_w = (const float*)d_in[16]; const float* ln2_b = (const float*)d_in[17];
  const float* Wfc = (const float*)d_in[18]; const float* bfc  = (const float*)d_in[19];
  const float* Wpr = (const float*)d_in[20]; const float* bpr  = (const float*)d_in[21];
  // d_in[22] = attn_dim = -2 (identity moveaxis) -> ignored

  char* ws = (char*)d_ws;
  const size_t MB = (size_t)1 << 20;
  unsigned short* hs_ln = (unsigned short*)(ws + 0 * MB);    // 16 MB, later LN2 out
  unsigned short* hp_ln = (unsigned short*)(ws + 16 * MB);   // 16 MB
  unsigned short* qb    = (unsigned short*)(ws + 32 * MB);   // 16 MB
  unsigned short* kb    = (unsigned short*)(ws + 48 * MB);   // 16 MB
  unsigned short* vtb   = (unsigned short*)(ws + 64 * MB);   // 16 MB, V^T [B,H,HD,L]
  unsigned short* tb    = (unsigned short*)(ws + 80 * MB);   // 16 MB (Wv temp, then attn Y)
  unsigned short* hb    = (unsigned short*)(ws + 96 * MB);   // 64 MB MLP hidden
  unsigned short* wqb   = (unsigned short*)(ws + 160 * MB);  // transposed bf16 weights
  unsigned short* wkb   = (unsigned short*)(ws + 162 * MB);
  unsigned short* wvb   = (unsigned short*)(ws + 164 * MB);
  unsigned short* wvtb  = (unsigned short*)(ws + 166 * MB);
  unsigned short* wcpb  = (unsigned short*)(ws + 168 * MB);
  unsigned short* wfcb  = (unsigned short*)(ws + 170 * MB);  // 8 MB
  unsigned short* wprb  = (unsigned short*)(ws + 178 * MB);  // 8 MB

  float* out_hp = (float*)d_out;
  dim3 blk(256);
  const int nCCCC = CC * CC, nCF = CC * FCDIM;

  cvt_f32_bf16_t<<<(nCCCC + 255) / 256, blk, 0, stream>>>(Wq,  wqb,  CC, CC);
  cvt_f32_bf16_t<<<(nCCCC + 255) / 256, blk, 0, stream>>>(Wk,  wkb,  CC, CC);
  cvt_f32_bf16_t<<<(nCCCC + 255) / 256, blk, 0, stream>>>(Wv,  wvb,  CC, CC);
  cvt_f32_bf16_t<<<(nCCCC + 255) / 256, blk, 0, stream>>>(Wvt, wvtb, CC, CC);
  cvt_f32_bf16_t<<<(nCCCC + 255) / 256, blk, 0, stream>>>(Wcp, wcpb, CC, CC);
  cvt_f32_bf16_t<<<(nCF + 255) / 256,  blk, 0, stream>>>(Wfc, wfcb, CC, FCDIM);
  cvt_f32_bf16_t<<<(nCF + 255) / 256,  blk, 0, stream>>>(Wpr, wprb, FCDIM, CC);

  layernorm_to_bf16<<<MM, blk, 0, stream>>>(h_private, ln1p_w, ln1p_b, hp_ln);
  layernorm_to_bf16<<<MM, blk, 0, stream>>>(h_shared,  ln1s_w, ln1s_b, hs_ln);

  dim3 gSq(CC / 256, MM / 64);
  gemm_bf16_wmma<0><<<gSq, blk, 0, stream>>>(hs_ln, wqb,  bq,  nullptr, qb,  nullptr, MM, CC, CC);
  gemm_bf16_wmma<0><<<gSq, blk, 0, stream>>>(hs_ln, wkb,  bk,  nullptr, kb,  nullptr, MM, CC, CC);
  gemm_bf16_wmma<0><<<gSq, blk, 0, stream>>>(hp_ln, wvb,  bv,  nullptr, tb,  nullptr, MM, CC, CC);
  gemm_bf16_wmma<3><<<gSq, blk, 0, stream>>>(tb,    wvtb, bvt, nullptr, vtb, nullptr, MM, CC, CC);

  attn_wmma<<<dim3(LL / 64, BB * HH), blk, 0, stream>>>(qb, kb, vtb, tb);

  // h_private = h_private + Y @ Wcp + bcp  (f32, into d_out)
  gemm_bf16_wmma<2><<<gSq, blk, 0, stream>>>(tb, wcpb, bcp, h_private, nullptr, out_hp, MM, CC, CC);

  layernorm_to_bf16<<<MM, blk, 0, stream>>>(out_hp, ln2_w, ln2_b, hs_ln);

  gemm_bf16_wmma<1><<<dim3(FCDIM / 256, MM / 64), blk, 0, stream>>>(
      hs_ln, wfcb, bfc, nullptr, hb, nullptr, MM, FCDIM, CC);
  gemm_bf16_wmma<2><<<gSq, blk, 0, stream>>>(
      hb, wprb, bpr, out_hp, nullptr, out_hp, MM, CC, FCDIM);

  // second output: pass-through h_shared
  hipMemcpyAsync(out_hp + (size_t)MM * CC, h_shared, (size_t)MM * CC * sizeof(float),
                 hipMemcpyDeviceToDevice, stream);
}
